// GNNModel_53120155517254
// MI455X (gfx1250) — matrix operand order repaired and checked by reading, not compiled, not run on hardware
//
#include <hip/hip_runtime.h>
#include <math.h>

#define NATOMS 50000
#define NE     1600000
#define HD     128
#define FD     64
#define GD     50
#define LD     3
#define NGR    256
#define CUTOFF_R 10.0f

typedef __attribute__((ext_vector_type(16))) __bf16 v16bf;
typedef __attribute__((ext_vector_type(8)))  float  v8f;

// ---- weight-fragment table (each frag = 32 lanes x 16 bf16 = 512 bf16) ----
#define FR_MW1 0     // L * 2 chunks * 4 tiles = 24
#define FR_MW2 24    // 24
#define FR_L1W 48    // L * 4 * 4 = 48
#define FR_L2W 96    // L * 2 * 8 = 48
#define FR_LW  144   // L * 4 * 8 = 96
#define FR_OW1 240   // 4 * 4 = 16
#define FR_TOTAL 256

__device__ __forceinline__ v8f wmma_bf16(v16bf a, v16bf b, v8f c) {
  return __builtin_amdgcn_wmma_f32_16x16x32_bf16(false, a, false, b, (short)0, c,
                                                 false, false);
}

__device__ __forceinline__ float sspf(float x) {  // shifted softplus
  float sp = (x > 20.0f) ? x : log1pf(__expf(x));
  return sp - 0.69314718055994531f;
}

// A-matrix (16x32, 16-bit) per-lane K index for element j (pairs per VGPR):
// lanes 0-15: V0..3 -> K 0..7, V4..7 -> K 16..23 ; lanes 16-31: +8
__device__ __forceinline__ int a_klocal(int lane, int j) {
  int v = j >> 1, p = j & 1;
  return 2 * v + ((v >= 4) ? 8 : 0) + ((lane >= 16) ? 8 : 0) + p;
}

__device__ __forceinline__ v16bf load_b_frag(const __bf16* __restrict__ wf,
                                             int frag, int lane) {
  return *(const v16bf*)(wf + (size_t)frag * 512 + lane * 16);
}

// Build A fragment from fp32 row-major source [rows x ld], 16 rows, one K-chunk
__device__ __forceinline__ v16bf load_a_f32(const float* __restrict__ src, int ld,
                                            int rowBase, int kBase, int kMax,
                                            int lane) {
  v16bf a;
  const float* row = src + (size_t)(rowBase + (lane & 15)) * ld;
#pragma unroll
  for (int j = 0; j < 16; ++j) {
    int k = kBase + a_klocal(lane, j);
    a[j] = (__bf16)((k < kMax) ? row[k] : 0.0f);
  }
  return a;
}

// Build A fragment from bf16 LDS tile [16 x ld]
__device__ __forceinline__ v16bf load_a_lds(const __bf16* lds, int ld, int kBase,
                                            int lane) {
  v16bf a;
  const __bf16* row = lds + (lane & 15) * ld;
#pragma unroll
  for (int j = 0; j < 16; ++j) a[j] = row[kBase + a_klocal(lane, j)];
  return a;
}

// ---------------- prep: swizzle fp32 weights into bf16 B-fragments ----------
__global__ void prep_weights(const float* __restrict__ mw1,
                             const float* __restrict__ mw2,
                             const float* __restrict__ l1w,
                             const float* __restrict__ l2w,
                             const float* __restrict__ lw,
                             const float* __restrict__ ow1,
                             __bf16* __restrict__ wf) {
  int gid = blockIdx.x * blockDim.x + threadIdx.x;
  int frag = gid >> 5, lane = gid & 31;
  if (frag >= FR_TOTAL) return;
  const float* src;
  int K, Ncols, c, t, f = frag;
  if (f < 24)        { int l = f / 8,  r = f % 8;  c = r / 4; t = r % 4; src = mw1 + (size_t)l * GD * FD; K = GD; Ncols = FD; }
  else if (f < 48)   { f -= 24;  int l = f / 8,  r = f % 8;  c = r / 4; t = r % 4; src = mw2 + (size_t)l * FD * FD; K = FD; Ncols = FD; }
  else if (f < 96)   { f -= 48;  int l = f / 16, r = f % 16; c = r / 4; t = r % 4; src = l1w + (size_t)l * HD * FD; K = HD; Ncols = FD; }
  else if (f < 144)  { f -= 96;  int l = f / 16, r = f % 16; c = r / 8; t = r % 8; src = l2w + (size_t)l * FD * HD; K = FD; Ncols = HD; }
  else if (f < 240)  { f -= 144; int l = f / 32, r = f % 32; c = r / 8; t = r % 8; src = lw  + (size_t)l * HD * HD; K = HD; Ncols = HD; }
  else               { f -= 240; c = f / 4; t = f % 4; src = ow1; K = HD; Ncols = HD / 2; }
  __bf16* out = wf + (size_t)frag * 512 + lane * 16;
#pragma unroll
  for (int j = 0; j < 16; ++j) {
    int v = j >> 1, p = j & 1;
    int k = c * 32 + 2 * v + p + ((v >= 4) ? 8 : 0) + ((lane >= 16) ? 8 : 0);  // B layout: same K order as A (a_klocal)
    int col = t * 16 + (lane & 15);
    out[j] = (__bf16)((k < K) ? src[(size_t)k * Ncols + col] : 0.0f);
  }
}

// ---------------- embedding gather ----------------
__global__ void embed_kernel(const int* __restrict__ z, const float* __restrict__ emb,
                             float* __restrict__ h) {
  int idx = blockIdx.x * blockDim.x + threadIdx.x;
  if (idx >= NATOMS * HD) return;
  int n = idx / HD, k = idx - n * HD;
  h[idx] = emb[(size_t)z[n] * HD + k];
}

// ---------------- edge geometry: distance + cosine cutoff ----------------
__global__ void edge_geom(const float* __restrict__ pos, const int* __restrict__ src,
                          const int* __restrict__ dst, float* __restrict__ dist,
                          float* __restrict__ cut) {
  int e = blockIdx.x * blockDim.x + threadIdx.x;
  if (e >= NE) return;
  int s = src[e], t = dst[e];
  float dx = pos[3 * s + 0] - pos[3 * t + 0];
  float dy = pos[3 * s + 1] - pos[3 * t + 1];
  float dz = pos[3 * s + 2] - pos[3 * t + 2];
  float d = sqrtf(dx * dx + dy * dy + dz * dz);
  dist[e] = d;
  cut[e] = 0.5f * (__cosf(d * (3.14159265358979323846f / CUTOFF_R)) + 1.0f);
}

__global__ void zero_kernel(float* __restrict__ p, size_t n) {
  size_t i = (size_t)blockIdx.x * blockDim.x + threadIdx.x;
  if (i < n) p[i] = 0.0f;
}

// ---------------- node projection: xj = h @ l1w[l]  (128 -> 64) ------------
__global__ void node_proj(const float* __restrict__ h, const __bf16* __restrict__ wf,
                          float* __restrict__ xj, int l) {
  int wid = threadIdx.x >> 5, lane = threadIdx.x & 31;
  int tile = blockIdx.x * 8 + wid;
  if (tile * 16 >= NATOMS) return;
  int rb = tile * 16;
  v8f acc[4] = {};
#pragma unroll
  for (int c = 0; c < 4; ++c) {
    v16bf a = load_a_f32(h, HD, rb, c * 32, HD, lane);
#pragma unroll
    for (int t = 0; t < 4; ++t) {
      v16bf b = load_b_frag(wf, FR_L1W + (l * 4 + c) * 4 + t, lane);
      acc[t] = wmma_bf16(a, b, acc[t]);
    }
  }
#pragma unroll
  for (int t = 0; t < 4; ++t)
#pragma unroll
    for (int r = 0; r < 8; ++r) {
      int m = r + ((lane >= 16) ? 8 : 0);
      int col = t * 16 + (lane & 15);
      xj[(size_t)(rb + m) * FD + col] = acc[t][r];
    }
}

// ------- fused edge kernel: gaussians -> MLP(WMMA) -> gather*W -> scatter ---
__global__ void edge_fused(const float* __restrict__ dist, const float* __restrict__ cut,
                           const int* __restrict__ srcI, const int* __restrict__ dstI,
                           const float* __restrict__ xj, const __bf16* __restrict__ wf,
                           const float* __restrict__ mb1, const float* __restrict__ mb2,
                           float* __restrict__ agg, int l) {
  __shared__ __bf16 stage[8][16 * FD];
  __shared__ float dsh[8][16], csh[8][16];
  __shared__ int ssh[8][16], tsh[8][16];
  int wid = threadIdx.x >> 5, lane = threadIdx.x & 31;
  long ebl = ((long)blockIdx.x * 8 + wid) * 16;
  if (ebl >= NE) return;
  int eb = (int)ebl;
  if (lane < 16) {
    int e = eb + lane;
    dsh[wid][lane] = dist[e];
    csh[wid][lane] = cut[e];
    ssh[wid][lane] = srcI[e];
    tsh[wid][lane] = dstI[e];
  }
  const float delta = CUTOFF_R / (GD - 1);
  const float coeff = -0.5f / (delta * delta);

  // stage 1: ssp(edge_attr @ mw1 + mb1), A built in-register from gaussians
  v8f acc1[4] = {};
#pragma unroll
  for (int c = 0; c < 2; ++c) {
    v16bf a;
    float dm = dsh[wid][lane & 15];
#pragma unroll
    for (int j = 0; j < 16; ++j) {
      int k = c * 32 + a_klocal(lane, j);
      float diff = dm - (float)k * delta;
      a[j] = (__bf16)((k < GD) ? __expf(coeff * diff * diff) : 0.0f);
    }
#pragma unroll
    for (int t = 0; t < 4; ++t) {
      v16bf b = load_b_frag(wf, FR_MW1 + (l * 2 + c) * 4 + t, lane);
      acc1[t] = wmma_bf16(a, b, acc1[t]);
    }
  }
#pragma unroll
  for (int t = 0; t < 4; ++t)
#pragma unroll
    for (int r = 0; r < 8; ++r) {
      int m = r + ((lane >= 16) ? 8 : 0);
      int col = t * 16 + (lane & 15);
      stage[wid][m * FD + col] = (__bf16)sspf(acc1[t][r] + mb1[l * FD + col]);
    }

  // stage 2: W = t1 @ mw2 + mb2
  v8f acc2[4] = {};
#pragma unroll
  for (int c = 0; c < 2; ++c) {
    v16bf a = load_a_lds(&stage[wid][0], FD, c * 32, lane);
#pragma unroll
    for (int t = 0; t < 4; ++t) {
      v16bf b = load_b_frag(wf, FR_MW2 + (l * 2 + c) * 4 + t, lane);
      acc2[t] = wmma_bf16(a, b, acc2[t]);
    }
  }

  // finish: W *= C ; msg = xj[src] * W ; atomic scatter into agg[dst]
#pragma unroll
  for (int t = 0; t < 4; ++t)
#pragma unroll
    for (int r = 0; r < 8; ++r) {
      int m = r + ((lane >= 16) ? 8 : 0);
      int col = t * 16 + (lane & 15);
      float W = (acc2[t][r] + mb2[l * FD + col]) * csh[wid][m];
      float msg = xj[(size_t)ssh[wid][m] * FD + col] * W;
      atomicAdd(&agg[(size_t)tsh[wid][m] * FD + col], msg);
    }
}

// ------- node update: h += (ssp(agg @ l2w + l2b)) @ lw + lb  ---------------
__global__ void node_update(const float* __restrict__ agg, const __bf16* __restrict__ wf,
                            const float* __restrict__ l2b, const float* __restrict__ lb,
                            float* __restrict__ h, int l) {
  __shared__ __bf16 st[8][16 * HD];
  int wid = threadIdx.x >> 5, lane = threadIdx.x & 31;
  int tile = blockIdx.x * 8 + wid;
  if (tile * 16 >= NATOMS) return;
  int rb = tile * 16;

  v8f acc1[8] = {};
#pragma unroll
  for (int c = 0; c < 2; ++c) {
    v16bf a = load_a_f32(agg, FD, rb, c * 32, FD, lane);
#pragma unroll
    for (int t = 0; t < 8; ++t) {
      v16bf b = load_b_frag(wf, FR_L2W + (l * 2 + c) * 8 + t, lane);
      acc1[t] = wmma_bf16(a, b, acc1[t]);
    }
  }
#pragma unroll
  for (int t = 0; t < 8; ++t)
#pragma unroll
    for (int r = 0; r < 8; ++r) {
      int m = r + ((lane >= 16) ? 8 : 0);
      int col = t * 16 + (lane & 15);
      st[wid][m * HD + col] = (__bf16)sspf(acc1[t][r] + l2b[l * HD + col]);
    }

  v8f acc2[8] = {};
#pragma unroll
  for (int c = 0; c < 4; ++c) {
    v16bf a = load_a_lds(&st[wid][0], HD, c * 32, lane);
#pragma unroll
    for (int t = 0; t < 8; ++t) {
      v16bf b = load_b_frag(wf, FR_LW + (l * 4 + c) * 8 + t, lane);
      acc2[t] = wmma_bf16(a, b, acc2[t]);
    }
  }
#pragma unroll
  for (int t = 0; t < 8; ++t)
#pragma unroll
    for (int r = 0; r < 8; ++r) {
      int m = r + ((lane >= 16) ? 8 : 0);
      int col = t * 16 + (lane & 15);
      size_t idx = (size_t)(rb + m) * HD + col;
      h[idx] = h[idx] + acc2[t][r] + lb[l * HD + col];
    }
}

// ------- readout: ssp(h@ow1+ob1) @ ow2 + ob2, segment-sum per graph --------
__global__ void readout(const float* __restrict__ h, const __bf16* __restrict__ wf,
                        const float* __restrict__ ob1, const float* __restrict__ ow2,
                        const float* __restrict__ ob2, const int* __restrict__ batch,
                        float* __restrict__ out) {
  int wid = threadIdx.x >> 5, lane = threadIdx.x & 31;
  int tile = blockIdx.x * 8 + wid;
  if (tile * 16 >= NATOMS) return;
  int rb = tile * 16;
  v8f acc[4] = {};
#pragma unroll
  for (int c = 0; c < 4; ++c) {
    v16bf a = load_a_f32(h, HD, rb, c * 32, HD, lane);
#pragma unroll
    for (int t = 0; t < 4; ++t) {
      v16bf b = load_b_frag(wf, FR_OW1 + c * 4 + t, lane);
      acc[t] = wmma_bf16(a, b, acc[t]);
    }
  }
  float part[8];
#pragma unroll
  for (int r = 0; r < 8; ++r) part[r] = 0.0f;
#pragma unroll
  for (int t = 0; t < 4; ++t)
#pragma unroll
    for (int r = 0; r < 8; ++r) {
      int col = t * 16 + (lane & 15);
      part[r] += sspf(acc[t][r] + ob1[col]) * ow2[col];
    }
#pragma unroll
  for (int r = 0; r < 8; ++r) {
#pragma unroll
    for (int off = 8; off >= 1; off >>= 1)
      part[r] += __shfl_xor(part[r], off, 16);   // reduce within 16-lane groups
  }
  if ((lane & 15) == 0) {
#pragma unroll
    for (int r = 0; r < 8; ++r) {
      int m = r + ((lane >= 16) ? 8 : 0);
      atomicAdd(&out[batch[rb + m]], part[r] + ob2[0]);
    }
  }
}

// ---------------------------------------------------------------------------
extern "C" void kernel_launch(void* const* d_in, const int* in_sizes, int n_in,
                              void* d_out, int out_size, void* d_ws, size_t ws_size,
                              hipStream_t stream) {
  (void)in_sizes; (void)n_in; (void)out_size; (void)ws_size;
  const int*   z     = (const int*)  d_in[0];
  const float* pos   = (const float*)d_in[1];
  const int*   batch = (const int*)  d_in[2];
  const int*   eidx  = (const int*)  d_in[3];
  const float* emb   = (const float*)d_in[4];
  const float* mw1   = (const float*)d_in[5];
  const float* mb1   = (const float*)d_in[6];
  const float* mw2   = (const float*)d_in[7];
  const float* mb2   = (const float*)d_in[8];
  const float* l1w   = (const float*)d_in[9];
  const float* l2w   = (const float*)d_in[10];
  const float* l2b   = (const float*)d_in[11];
  const float* lw    = (const float*)d_in[12];
  const float* lb    = (const float*)d_in[13];
  const float* ow1   = (const float*)d_in[14];
  const float* ob1   = (const float*)d_in[15];
  const float* ow2   = (const float*)d_in[16];
  const float* ob2   = (const float*)d_in[17];
  float* out = (float*)d_out;

  char* base = (char*)d_ws;
  size_t off = 0;
  auto take = [&](size_t bytes) -> char* {
    char* p = base + off;
    off = (off + bytes + 255) & ~(size_t)255;
    return p;
  };
  float*  h    = (float*) take((size_t)NATOMS * HD * 4);
  float*  xj   = (float*) take((size_t)NATOMS * FD * 4);
  float*  agg  = (float*) take((size_t)NATOMS * FD * 4);
  float*  dist = (float*) take((size_t)NE * 4);
  float*  cut  = (float*) take((size_t)NE * 4);
  __bf16* wf   = (__bf16*)take((size_t)FR_TOTAL * 512 * 2);

  const int* srcI = eidx;        // edge_index[0]
  const int* dstI = eidx + NE;   // edge_index[1]

  prep_weights<<<(FR_TOTAL * 32 + 255) / 256, 256, 0, stream>>>(mw1, mw2, l1w, l2w,
                                                               lw, ow1, wf);
  embed_kernel<<<(NATOMS * HD + 255) / 256, 256, 0, stream>>>(z, emb, h);
  edge_geom<<<(NE + 255) / 256, 256, 0, stream>>>(pos, srcI, dstI, dist, cut);

  const int nodeTiles = NATOMS / 16;               // 3125 exact
  const int nodeBlocks = (nodeTiles + 7) / 8;
  const int edgeBlocks = NE / 16 / 8;              // 12500 exact

  for (int l = 0; l < LD; ++l) {
    zero_kernel<<<(NATOMS * FD + 255) / 256, 256, 0, stream>>>(agg,
                                                               (size_t)NATOMS * FD);
    node_proj<<<nodeBlocks, 256, 0, stream>>>(h, wf, xj, l);
    edge_fused<<<edgeBlocks, 256, 0, stream>>>(dist, cut, srcI, dstI, xj, wf, mb1,
                                               mb2, agg, l);
    node_update<<<nodeBlocks, 256, 0, stream>>>(agg, wf, l2b, lb, h, l);
  }

  zero_kernel<<<1, 256, 0, stream>>>(out, (size_t)NGR);
  readout<<<nodeBlocks, 256, 0, stream>>>(h, wf, ob1, ow2, ob2, batch, out);
}
